// DynamicResolutionAttention_5729486373657
// MI455X (gfx1250) — compile-verified, matrix operand order checked by reading, not run on hardware
//
#include <hip/hip_runtime.h>

#define Bx 2
#define Tx 2048
#define Cx 1024
#define Hx 16
#define HDx 64
#define M_TOT (Bx * Tx)   // 4096

typedef __attribute__((ext_vector_type(16))) _Float16 v16h;
typedef __attribute__((ext_vector_type(8)))  _Float16 v8h;
typedef __attribute__((ext_vector_type(8)))  float    v8f;

// ---------- DPP row reductions (within 16-lane group, 1 instruction/stage) ----
// DPP16 ROW_XMASK applied directly as the src0 modifier of max/add:
//   x = op(dpp_xmask(x), x)  -- no extra movs, no canonicalize self-max.
__device__ __forceinline__ float rowmax16(float x) {
  asm("v_max_num_f32_dpp %0, %0, %0 row_xmask:1 row_mask:0xf bank_mask:0xf\n\t"
      "v_max_num_f32_dpp %0, %0, %0 row_xmask:2 row_mask:0xf bank_mask:0xf\n\t"
      "v_max_num_f32_dpp %0, %0, %0 row_xmask:4 row_mask:0xf bank_mask:0xf\n\t"
      "v_max_num_f32_dpp %0, %0, %0 row_xmask:8 row_mask:0xf bank_mask:0xf"
      : "+v"(x));
  return x;
}
__device__ __forceinline__ float rowsum16(float x) {
  asm("v_add_f32_dpp %0, %0, %0 row_xmask:1 row_mask:0xf bank_mask:0xf\n\t"
      "v_add_f32_dpp %0, %0, %0 row_xmask:2 row_mask:0xf bank_mask:0xf\n\t"
      "v_add_f32_dpp %0, %0, %0 row_xmask:4 row_mask:0xf bank_mask:0xf\n\t"
      "v_add_f32_dpp %0, %0, %0 row_xmask:8 row_mask:0xf bank_mask:0xf"
      : "+v"(x));
  return x;
}

// Load a 16x32 f16 WMMA A/B operand from a row-major matrix (leading dim ld).
// ISA 16-bit A-matrix 16x32 layout: lane l (0-15) = row l, holds k = {0..7,16..23};
// lane 16+l = row l, holds k = {8..15, 24..31}. B operand mirrors with row->col.
__device__ __forceinline__ v16h load_frag16(const _Float16* __restrict__ base, int ld) {
  int lane = threadIdx.x & 31;
  int row  = lane & 15;
  int h8   = (lane >> 4) & 1;
  const _Float16* p = base + row * ld + h8 * 8;
  v8h a0 = *(const v8h*)(p);
  v8h a1 = *(const v8h*)(p + 16);
  v16h r;
#pragma unroll
  for (int i = 0; i < 8; ++i) { r[i] = a0[i]; r[i + 8] = a1[i]; }
  return r;
}

// CDNA5 async global->LDS copy, 16B per lane, tracked by ASYNCcnt.
__device__ __forceinline__ void async_copy_b128(uint32_t lds_off, const void* gaddr) {
  asm volatile("global_load_async_to_lds_b128 %0, %1, off"
               :: "v"(lds_off), "v"((uint64_t)(uintptr_t)gaddr) : "memory");
}
__device__ __forceinline__ void wait_asynccnt0() {
  asm volatile("s_wait_asynccnt 0" ::: "memory");
}

__global__ void cvt_f32_to_f16(const float* __restrict__ in, _Float16* __restrict__ out, int n) {
  int i = (blockIdx.x * blockDim.x + threadIdx.x) * 8;
  if (i < n) {
    v8f v = *(const v8f*)(in + i);
    v8h h;
#pragma unroll
    for (int j = 0; j < 8; ++j) h[j] = (_Float16)v[j];
    *(v8h*)(out + i) = h;
  }
}

// MODE 0: f16 out as [B,H,T,HD] (Q,K)   MODE 1: f16 out as [B,H,HD,T] (V transposed)
// MODE 2: f32 out as [M,C] (final projection into d_out)
// One wave computes a 32x64 output tile; k-loop software-pipelined (ping-pong).
template <int MODE>
__global__ void __launch_bounds__(128)
proj_kernel(const _Float16* __restrict__ Xh,   // [M_TOT, C] f16
            const _Float16* __restrict__ Wh,   // [C, C] f16 row-major (d, c): y = x @ W^T
            const float* __restrict__ bias,    // [C]
            void* __restrict__ out) {
  int wave = (int)((blockIdx.x * blockDim.x + threadIdx.x) >> 5);
  int lane = threadIdx.x & 31;
  const int NW = Cx / 64;                 // 16 waves across N
  int m0 = (wave / NW) * 32;
  int n0 = (wave % NW) * 64;

  v8f acc[2][4] = {};
  v16h a0[2], b0[4], a1[2], b1[4];

  auto loadAB = [&](v16h* a, v16h* b, int k0) {
#pragma unroll
    for (int i = 0; i < 2; ++i)
      a[i] = load_frag16(Xh + (long)(m0 + i * 16) * Cx + k0, Cx);
#pragma unroll
    for (int j = 0; j < 4; ++j)
      b[j] = load_frag16(Wh + (long)(n0 + j * 16) * Cx + k0, Cx);
  };
  auto mma = [&](v16h* a, v16h* b) {
#pragma unroll
    for (int i = 0; i < 2; ++i)
#pragma unroll
      for (int j = 0; j < 4; ++j)
        acc[i][j] = __builtin_amdgcn_wmma_f32_16x16x32_f16(false, a[i], false, b[j],
                                                           (short)0, acc[i][j], false, false);
  };

  loadAB(a0, b0, 0);
  for (int k0 = 0; k0 < Cx; k0 += 64) {
    loadAB(a1, b1, k0 + 32);            // in flight while first 8 WMMAs run
    mma(a0, b0);
    if (k0 + 64 < Cx) loadAB(a0, b0, k0 + 64);
    mma(a1, b1);
  }

  // C/D layout: VGPR r -> row (+8 for lanes 16-31), col = (lane&15)
  int mrow = (lane < 16) ? 0 : 8;
#pragma unroll
  for (int i = 0; i < 2; ++i) {
#pragma unroll
    for (int j = 0; j < 4; ++j) {
      int n = n0 + j * 16 + (lane & 15);
      float bv = bias[n];
#pragma unroll
      for (int r = 0; r < 8; ++r) {
        int m = m0 + i * 16 + r + mrow;
        float val = acc[i][j][r] + bv;
        if (MODE == 2) {
          ((float*)out)[(long)m * Cx + n] = val;
        } else {
          int b = m / Tx, t = m % Tx;
          int h = n / HDx, hd = n % HDx;
          long idx = (MODE == 0)
                   ? (((long)(b * Hx + h) * Tx + t) * HDx + hd)
                   : (((long)(b * Hx + h) * HDx + hd) * Tx + t);
          ((_Float16*)out)[idx] = (_Float16)val;
        }
      }
    }
  }
}

// Flash attention. 4 waves/block, one 16-query tile each, all sharing one
// (batch,head): K/V 32-key blocks are cooperatively staged into double-buffered
// LDS with async global->LDS copies (ASYNCcnt), overlapped with WMMA compute.
// Softmax runs in the exp2 domain (log2e folded into the Q scale) so the
// transcendentals are raw v_exp_f32.
__global__ void __launch_bounds__(128)
attn_kernel(const _Float16* __restrict__ Qh,   // [B,H,T,HD]
            const _Float16* __restrict__ Kh,   // [B,H,T,HD]
            const _Float16* __restrict__ Vt,   // [B,H,HD,T]
            const float* __restrict__ resolve,
            _Float16* __restrict__ Yh) {       // [B*T, C]
  // Staging: K tile 32x64 f16 (4KB) + V^T tile 64x32 f16 (4KB), double buffered.
  __shared__ __attribute__((aligned(16))) _Float16 kvbuf[2][2 * 32 * 64];
  __shared__ __attribute__((aligned(16))) _Float16 plds[4][16 * 32];  // per-wave P tile
  int tid  = threadIdx.x;
  int wib  = tid >> 5;
  int lane = tid & 31;
  const int QT = Tx / 16;                      // 128 q-tiles per head
  int bh = (blockIdx.x * 4) / QT;              // same for all 4 waves in block
  int q0 = ((blockIdx.x * 4 + wib) % QT) * 16;
  const _Float16* Qbase = Qh + ((long)bh * Tx + q0) * HDx;
  const char*     Kg    = (const char*)(Kh + (long)bh * Tx * HDx);
  const char*     Vg    = (const char*)(Vt + (long)bh * HDx * Tx);
  _Float16* myp = plds[wib];

  // 1/sqrt(HD) * temperature * log2(e): softmax computed with exp2.
  float sc = 0.125f * (0.5f + 0.5f * resolve[0]) * 1.4426950408889634f;

  v16h aq0 = load_frag16(Qbase + 0,  HDx);
  v16h aq1 = load_frag16(Qbase + 32, HDx);
  {   // fold the scale into Q once (f16): S comes out of WMMA pre-scaled
    _Float16 hsc = (_Float16)sc;
#pragma unroll
    for (int i = 0; i < 16; ++i) { aq0[i] *= hsc; aq1[i] *= hsc; }
  }

  float rowM[8], rowL[8];
#pragma unroll
  for (int r = 0; r < 8; ++r) { rowM[r] = -1e30f; rowL[r] = 0.0f; }
  v8f o[4] = {};

  // Cooperative async stage of one 32-key block into kvbuf[buf].
  // K tile: 4096B contiguous at Kg + key0*128. V tile: 64 rows of 64B, stride 2*T.
  auto issue_stage = [&](int buf, int key0) {
    uint32_t ldsK = (uint32_t)(uintptr_t)&kvbuf[buf][0];
    uint32_t ldsV = (uint32_t)(uintptr_t)&kvbuf[buf][32 * 64];
#pragma unroll
    for (int c = 0; c < 2; ++c) {               // 256 chunks over 128 threads
      int ck = tid + c * 128;
      async_copy_b128(ldsK + ck * 16, Kg + (long)key0 * 128 + ck * 16);
      int row = ck >> 2, sub = ck & 3;
      async_copy_b128(ldsV + row * 64 + sub * 16,
                      Vg + (long)row * (Tx * 2) + (long)key0 * 2 + sub * 16);
    }
  };

  auto process = [&](int buf) {
    const _Float16* ldsK = &kvbuf[buf][0];        // [32][64]
    const _Float16* ldsV = &kvbuf[buf][32 * 64];  // [64][32]
    v16h kb0 = load_frag16(ldsK + 0 * 64 + 0,  64);
    v16h kb1 = load_frag16(ldsK + 0 * 64 + 32, 64);
    v16h kb2 = load_frag16(ldsK + 16 * 64 + 0,  64);
    v16h kb3 = load_frag16(ldsK + 16 * 64 + 32, 64);

    v8f s0 = {}, s1 = {};
    s0 = __builtin_amdgcn_wmma_f32_16x16x32_f16(false, aq0, false, kb0, (short)0, s0, false, false);
    s0 = __builtin_amdgcn_wmma_f32_16x16x32_f16(false, aq1, false, kb1, (short)0, s0, false, false);
    s1 = __builtin_amdgcn_wmma_f32_16x16x32_f16(false, aq0, false, kb2, (short)0, s1, false, false);
    s1 = __builtin_amdgcn_wmma_f32_16x16x32_f16(false, aq1, false, kb3, (short)0, s1, false, false);

    // Online softmax (exp2 domain): DPP ROW_XMASK butterflies, raw v_exp_f32.
#pragma unroll
    for (int r = 0; r < 8; ++r) {
      float a0 = s0[r], a1 = s1[r];
      float mx = rowmax16(fmaxf(a0, a1));
      float mnew = fmaxf(rowM[r], mx);
      float corr = __builtin_amdgcn_exp2f(rowM[r] - mnew);
      float p0 = __builtin_amdgcn_exp2f(a0 - mnew);
      float p1 = __builtin_amdgcn_exp2f(a1 - mnew);
      float rs = rowsum16(p0 + p1);
      rowL[r] = rowL[r] * corr + rs;
      rowM[r] = mnew;
      s0[r] = p0; s1[r] = p1;
#pragma unroll
      for (int j = 0; j < 4; ++j) o[j][r] *= corr;
    }

    // Relayout P (C/D layout) -> A-operand layout via per-wave LDS tile.
    {
      int mrow = (lane < 16) ? 0 : 8;
      int col  = lane & 15;
#pragma unroll
      for (int r = 0; r < 8; ++r) {
        myp[(mrow + r) * 32 + col]      = (_Float16)s0[r];
        myp[(mrow + r) * 32 + 16 + col] = (_Float16)s1[r];
      }
    }
    v16h pa = load_frag16(myp, 32);

#pragma unroll
    for (int j = 0; j < 4; ++j) {
      v16h vb = load_frag16(ldsV + (long)(j * 16) * 32, 32);
      o[j] = __builtin_amdgcn_wmma_f32_16x16x32_f16(false, pa, false, vb, (short)0, o[j], false, false);
    }
  };

  issue_stage(0, 0);
  wait_asynccnt0();
  __syncthreads();
  const int NBLK = Tx / 32;                     // 64
  for (int it = 0; it < NBLK; ++it) {
    if (it + 1 < NBLK) issue_stage((it + 1) & 1, (it + 1) * 32);  // overlaps compute
    process(it & 1);
    wait_asynccnt0();      // own-wave async copies landed
    __syncthreads();       // cross-wave visibility; also guards the WAR on buffers
  }

  // Normalize and write f16 attention output [B*T, C] with d = h*HD + hd.
  int b = bh / Hx, h = bh % Hx;
  int mrow = (lane < 16) ? 0 : 8;
#pragma unroll
  for (int j = 0; j < 4; ++j) {
    int d = h * HDx + j * 16 + (lane & 15);
#pragma unroll
    for (int r = 0; r < 8; ++r) {
      int t = q0 + r + mrow;
      float val = o[j][r] / rowL[r];
      Yh[((long)(b * Tx + t)) * Cx + d] = (_Float16)val;
    }
  }
}

extern "C" void kernel_launch(void* const* d_in, const int* in_sizes, int n_in,
                              void* d_out, int out_size, void* d_ws, size_t ws_size,
                              hipStream_t stream) {
  const float* x   = (const float*)d_in[0];
  const float* Wq  = (const float*)d_in[1];
  const float* bq  = (const float*)d_in[2];
  const float* Wk  = (const float*)d_in[3];
  const float* bk  = (const float*)d_in[4];
  const float* Wv  = (const float*)d_in[5];
  const float* bv  = (const float*)d_in[6];
  const float* Wp  = (const float*)d_in[7];
  const float* bp  = (const float*)d_in[8];
  const float* res = (const float*)d_in[9];
  float* out = (float*)d_out;

  // Workspace layout (bytes), 48 MiB total
  char* ws = (char*)d_ws;
  const size_t XH  = 0;
  const size_t WQH = XH  + (size_t)M_TOT * Cx * 2;
  const size_t WKH = WQH + (size_t)Cx * Cx * 2;
  const size_t WVH = WKH + (size_t)Cx * Cx * 2;
  const size_t WPH = WVH + (size_t)Cx * Cx * 2;
  const size_t QH  = WPH + (size_t)Cx * Cx * 2;
  const size_t KH  = QH  + (size_t)M_TOT * Cx * 2;
  const size_t VT  = KH  + (size_t)M_TOT * Cx * 2;
  const size_t YH  = VT  + (size_t)M_TOT * Cx * 2;
  _Float16* xh  = (_Float16*)(ws + XH);
  _Float16* wqh = (_Float16*)(ws + WQH);
  _Float16* wkh = (_Float16*)(ws + WKH);
  _Float16* wvh = (_Float16*)(ws + WVH);
  _Float16* wph = (_Float16*)(ws + WPH);
  _Float16* qh  = (_Float16*)(ws + QH);
  _Float16* kh  = (_Float16*)(ws + KH);
  _Float16* vt  = (_Float16*)(ws + VT);
  _Float16* yh  = (_Float16*)(ws + YH);

  // 1) f32 -> f16 conversions (8 elems/thread, b128 both sides)
  {
    int n = M_TOT * Cx;
    cvt_f32_to_f16<<<n / (256 * 8), 256, 0, stream>>>(x, xh, n);
    n = Cx * Cx;
    cvt_f32_to_f16<<<n / (256 * 8), 256, 0, stream>>>(Wq, wqh, n);
    cvt_f32_to_f16<<<n / (256 * 8), 256, 0, stream>>>(Wk, wkh, n);
    cvt_f32_to_f16<<<n / (256 * 8), 256, 0, stream>>>(Wv, wvh, n);
    cvt_f32_to_f16<<<n / (256 * 8), 256, 0, stream>>>(Wp, wph, n);
  }

  // 2) QKV projections: 32x64 tile/wave -> 2048 waves -> 512 blocks x 128 threads
  const int PROJ_BLOCKS = (M_TOT / 32) * (Cx / 64) / 4;  // 512
  proj_kernel<0><<<PROJ_BLOCKS, 128, 0, stream>>>(xh, wqh, bq, (void*)qh);
  proj_kernel<0><<<PROJ_BLOCKS, 128, 0, stream>>>(xh, wkh, bk, (void*)kh);
  proj_kernel<1><<<PROJ_BLOCKS, 128, 0, stream>>>(xh, wvh, bv, (void*)vt);

  // 3) Flash attention: B*H*(T/16) = 4096 waves -> 1024 blocks x 128 threads
  const int ATTN_BLOCKS = Bx * Hx * (Tx / 16) / 4;       // 1024
  attn_kernel<<<ATTN_BLOCKS, 128, 0, stream>>>(qh, kh, vt, res, yh);

  // 4) Output projection (fp32 result + bias) into d_out
  proj_kernel<2><<<PROJ_BLOCKS, 128, 0, stream>>>(yh, wph, bp, (void*)out);
  (void)in_sizes; (void)n_in; (void)out_size; (void)ws_size;
}
